// Seq2SeqNet_34943853920752
// MI455X (gfx1250) — matrix operand order repaired and checked by reading, not compiled, not run on hardware
//
#include <hip/hip_runtime.h>
#include <hip/hip_bf16.h>

// ---------------------------------------------------------------------------
// Seq2Seq: encoder GEMM -> (hoisted) LSTM input GEMM -> persistent-kernel
//          recurrence (64 waves, atomic device barrier per step, W_hh^T
//          resident in LDS) -> fused 2-layer MLP.
// All GEMMs on v_wmma_f32_16x16x32_bf16 (bf16 A/B, f32 accumulate).
// Sizes: B=32 S=1024 D_IN=256 E=512 H=512 F=2048 O=512.  Workspace ~177 MB.
// d_out layout: output[S,B,H] | h_n[B,H] | c_n[B,H] | embed_out[S,B,O]
// ---------------------------------------------------------------------------

typedef __attribute__((ext_vector_type(16))) __bf16 v16bf;
typedef __attribute__((ext_vector_type(8)))  float  v8f;

union BF16x16 { uint4 u[2]; unsigned short s[16]; v16bf v; };

__device__ __forceinline__ unsigned short f2bf(float f) {
  unsigned u = __builtin_bit_cast(unsigned, f);
  u += 0x7FFFu + ((u >> 16) & 1u);            // round-to-nearest-even
  return (unsigned short)(u >> 16);
}
__device__ __forceinline__ float bf2f(unsigned short s) {
  unsigned u = ((unsigned)s) << 16;
  return __builtin_bit_cast(float, u);
}

// 16 contiguous halves -> v16bf (works for global or LDS pointers)
__device__ __forceinline__ v16bf load16(const unsigned short* p) {
  BF16x16 x;
  x.u[0] = *(const uint4*)p;
  x.u[1] = *(const uint4*)(p + 8);
  return x.v;
}

// A-operand tile (16x32 bf16, row-major src, ld elements).
// Contiguous k order (same as load_b_bf16): lane<16 holds row (lane), K k0..k0+15;
//             lane>=16 holds row (lane-16), K k0+16..k0+31.
__device__ __forceinline__ v16bf load_a_bf16(const unsigned short* base,
                                             int row0, int k0, int ld, int lane) {
  const unsigned short* p =
      base + (size_t)(row0 + (lane & 15)) * ld + k0 + ((lane >> 4) << 4);
  BF16x16 x;
  x.u[0] = *(const uint4*)p;          // K +0..7
  x.u[1] = *(const uint4*)(p + 8);    // K +8..15
  return x.v;
}

// B-operand tile (32x16 bf16) from transposed weights Wt[N][K] (ld = K).
// lane<16 = column lane, K k0..k0+15; lane>=16 = column lane-16, K k0+16..+31.
__device__ __forceinline__ v16bf load_b_bf16(const unsigned short* baseT,
                                             int n0, int k0, int ld, int lane) {
  return load16(baseT + (size_t)(n0 + (lane & 15)) * ld + k0 + ((lane >> 4) << 4));
}

// A-operand tile from fp32 source (converted on the fly).
__device__ __forceinline__ v16bf load_a_f32(const float* base, int row0, int k0,
                                            int ld, int lane) {
  const float* p = base + (size_t)(row0 + (lane & 15)) * ld + k0 + ((lane >> 4) << 4);
  BF16x16 x;
#pragma unroll
  for (int i = 0; i < 8; ++i) {
    x.s[i]     = f2bf(p[i]);          // K +0..7
    x.s[8 + i] = f2bf(p[8 + i]);      // K +8..15
  }
  return x.v;
}

__device__ __forceinline__ v16bf relu_v(v16bf v) {
  BF16x16 x; x.v = v;
#pragma unroll
  for (int i = 0; i < 16; ++i)
    x.s[i] = (x.s[i] & 0x8000u) ? (unsigned short)0 : x.s[i];
  return x.v;
}

__device__ __forceinline__ v8f wmma_bf16(v16bf a, v16bf b, v8f c) {
  return __builtin_amdgcn_wmma_f32_16x16x32_bf16(false, a, false, b,
                                                 (short)0, c, false, false);
}

__device__ __forceinline__ float sigmoidf_(float x) {
  return 1.0f / (1.0f + __expf(-x));
}

// ---------------------------------------------------------------------------
// Weight transpose + fp32->bf16 convert: dst[n*K+k] = bf16(src[k*N+n])
// ---------------------------------------------------------------------------
__global__ void k_transpose_cvt(const float* __restrict__ src,
                                unsigned short* __restrict__ dst, int K, int N) {
  int idx = blockIdx.x * blockDim.x + threadIdx.x;
  if (idx >= K * N) return;
  int k = idx / N, n = idx - k * N;
  dst[(size_t)n * K + k] = f2bf(src[idx]);
}

__global__ void k_init_state(unsigned short* h0, unsigned short* h1,
                             unsigned int* cnt) {
  int i = blockIdx.x * blockDim.x + threadIdx.x;
  if (i < 32 * 512) { h0[i] = 0; h1[i] = 0; }
  if (i == 0) *cnt = 0u;          // step-barrier counter, reset every call
}

// ---------------------------------------------------------------------------
// Encoder: emb[s*32+b][e] = bf16( x[b*1024+s][:] @ W_enc + b_enc )
// grid (1, 2048) x 256 threads; each wave: 16 rows x 64 cols, K=256.
// ---------------------------------------------------------------------------
__global__ void k_encoder(const float* __restrict__ x,
                          const unsigned short* __restrict__ wencT,
                          const float* __restrict__ b_enc,
                          unsigned short* __restrict__ emb) {
  const int lane = threadIdx.x & 31;
  const int wv   = threadIdx.x >> 5;
  const int r0   = blockIdx.y * 16;      // row tile of [B*S] (b-major)
  const int n0   = wv * 64;
  v8f acc[4];
#pragma unroll
  for (int t = 0; t < 4; ++t) {
    float bias = b_enc[n0 + t * 16 + (lane & 15)];
#pragma unroll
    for (int e = 0; e < 8; ++e) acc[t][e] = bias;
  }
  for (int k0 = 0; k0 < 256; k0 += 32) {
    v16bf a = load_a_f32(x, r0, k0, 256, lane);
#pragma unroll
    for (int t = 0; t < 4; ++t) {
      v16bf b = load_b_bf16(wencT, n0 + t * 16, k0, 256, lane);
      acc[t] = wmma_bf16(a, b, acc[t]);
    }
  }
  const int bidx = r0 >> 10;             // batch index
  const int s0   = r0 & 1023;            // sequence base
#pragma unroll
  for (int t = 0; t < 4; ++t) {
    int col = n0 + t * 16 + (lane & 15);
#pragma unroll
    for (int e = 0; e < 8; ++e) {
      int m = e + ((lane >> 4) << 3);    // local row 0..15
      emb[(size_t)((s0 + m) * 32 + bidx) * 512 + col] = f2bf(acc[t][e]);
    }
  }
}

// ---------------------------------------------------------------------------
// Hoisted LSTM input projection: zx[r][n] = bf16( emb[r][:] @ W_ih + b_lstm )
// grid (4, 2048) x 256; wave: 16 rows x 64 cols, K=512.
// ---------------------------------------------------------------------------
__global__ void k_zx(const unsigned short* __restrict__ emb,
                     const unsigned short* __restrict__ wihT,
                     const float* __restrict__ b_lstm,
                     unsigned short* __restrict__ zx) {
  const int lane = threadIdx.x & 31;
  const int wv   = threadIdx.x >> 5;
  const int r0   = blockIdx.y * 16;
  const int n0   = blockIdx.x * 512 + wv * 64;
  v8f acc[4];
#pragma unroll
  for (int t = 0; t < 4; ++t) {
    float bias = b_lstm[n0 + t * 16 + (lane & 15)];
#pragma unroll
    for (int e = 0; e < 8; ++e) acc[t][e] = bias;
  }
  for (int k0 = 0; k0 < 512; k0 += 32) {
    v16bf a = load_a_bf16(emb, r0, k0, 512, lane);
#pragma unroll
    for (int t = 0; t < 4; ++t) {
      v16bf b = load_b_bf16(wihT, n0 + t * 16, k0, 512, lane);
      acc[t] = wmma_bf16(a, b, acc[t]);
    }
  }
#pragma unroll
  for (int t = 0; t < 4; ++t) {
    int col = n0 + t * 16 + (lane & 15);
#pragma unroll
    for (int e = 0; e < 8; ++e) {
      int m = e + ((lane >> 4) << 3);
      zx[(size_t)(r0 + m) * 2048 + col] = f2bf(acc[t][e]);
    }
  }
}

// ---------------------------------------------------------------------------
// Persistent LSTM recurrence.  grid 64 x 32 threads (one wave per WG).
// Wave (m0,j0) owns one 16x16 h-tile for all 1024 steps:
//   - W_hh^T panel (4 gates x 16 cols x K=512 = 64 KB) staged ONCE into LDS,
//     pre-packed in exact B-operand lane order -> 2 conflict-free ds_load_b128
//     per tile in the hot loop.
//   - c kept in registers for the whole sequence.
//   - per-step device barrier: release fetch_add + acquire spin (s_sleep).
// ---------------------------------------------------------------------------
__global__ void __launch_bounds__(32)
k_lstm_persistent(const unsigned short* __restrict__ zx,
                  const unsigned short* __restrict__ whhT,
                  unsigned short* __restrict__ h0,
                  unsigned short* __restrict__ h1,
                  float* __restrict__ out,
                  unsigned int* __restrict__ cnt) {
  __shared__ unsigned short bpanel[64 * 512];   // 64 KB
  const int lane = threadIdx.x;                 // block == one wave
  const int m0   = (blockIdx.x >> 5) << 4;      // 0 or 16
  const int j0   = (blockIdx.x & 31) << 4;      // 0..496

  // Stage W_hh^T panel: block bi = gate*16 + ktile; lane L owns 32 bytes.
  for (int bi = 0; bi < 64; ++bi) {
    int g = bi >> 4, kt = bi & 15;
    const unsigned short* src = whhT +
        (size_t)(g * 512 + j0 + (lane & 15)) * 512 + kt * 32 + ((lane >> 4) << 4);
    uint4 lo = *(const uint4*)src;
    uint4 hi = *(const uint4*)(src + 8);
    *(uint4*)&bpanel[bi * 512 + lane * 16]     = lo;
    *(uint4*)&bpanel[bi * 512 + lane * 16 + 8] = hi;
  }
  __syncthreads();

  const int j = j0 + (lane & 15);
  float c8[8];
#pragma unroll
  for (int e = 0; e < 8; ++e) c8[e] = 0.0f;

  for (int t = 0; t < 1024; ++t) {
    const unsigned short* hp = (t & 1) ? h1 : h0;
    unsigned short*       hn = (t & 1) ? h0 : h1;
    const unsigned short* zrow = zx + (size_t)t * 32 * 2048;

    v8f acc[4];                                  // gates i,f,g,o
#pragma unroll
    for (int g = 0; g < 4; ++g) {
      int col = g * 512 + j;
#pragma unroll
      for (int e = 0; e < 8; ++e)
        acc[g][e] = bf2f(zrow[(size_t)(m0 + e + ((lane >> 4) << 3)) * 2048 + col]);
    }
    if (t < 1023) {        // pull step t+1 pre-activations toward the caches
#pragma unroll
      for (int g = 0; g < 4; ++g)
        __builtin_prefetch(zrow + 32 * 2048 +
                           (size_t)(m0 + ((lane >> 4) << 3)) * 2048 + g * 512 + j,
                           0, 1);
    }
    for (int kt = 0; kt < 16; ++kt) {
      v16bf a = load_a_bf16(hp, m0, kt * 32, 512, lane);
#pragma unroll
      for (int g = 0; g < 4; ++g) {
        v16bf b = load16(&bpanel[(g * 16 + kt) * 512 + lane * 16]);
        acc[g] = wmma_bf16(a, b, acc[g]);
      }
    }
#pragma unroll
    for (int e = 0; e < 8; ++e) {
      int m = m0 + e + ((lane >> 4) << 3);
      float ig = sigmoidf_(acc[0][e]);
      float fg = sigmoidf_(acc[1][e]);
      float gg = tanhf(acc[2][e]);
      float og = sigmoidf_(acc[3][e]);
      float cn = fg * c8[e] + ig * gg;           // (m,j) owned by this lane
      c8[e] = cn;
      float h = og * tanhf(cn);
      int idx = m * 512 + j;
      __builtin_nontemporal_store(h, &out[(size_t)t * 16384 + idx]);
      hn[idx] = f2bf(h);
      if (t == 1023) {
        out[16777216 + idx]         = h;         // h_n
        out[16777216 + 16384 + idx] = cn;        // c_n
      }
    }
    if (t < 1023) {                              // device-scope step barrier
      if (lane == 0)
        __hip_atomic_fetch_add(cnt, 1u, __ATOMIC_RELEASE,
                               __HIP_MEMORY_SCOPE_AGENT);
      unsigned int target = 64u * (unsigned)(t + 1);
      while (__hip_atomic_load(cnt, __ATOMIC_ACQUIRE,
                               __HIP_MEMORY_SCOPE_AGENT) < target)
        __builtin_amdgcn_s_sleep(1);
    }
  }
}

// ---------------------------------------------------------------------------
// Fused MLP: out = (relu(emb) @ W1 + b1) @ W2 + b2, hidden staged in LDS.
// grid 2048 x 256 threads; WG owns 16 rows; hidden [16][2048] bf16 = 64 KB LDS.
// ---------------------------------------------------------------------------
__global__ void k_mlp(const unsigned short* __restrict__ emb,
                      const unsigned short* __restrict__ w1T,
                      const float* __restrict__ b1,
                      const unsigned short* __restrict__ w2T,
                      const float* __restrict__ b2,
                      float* __restrict__ out_embed) {
  __shared__ unsigned short hid[16 * 2048];
  const int lane = threadIdx.x & 31;
  const int wv   = threadIdx.x >> 5;
  const int r0   = blockIdx.x * 16;
  // ---- phase 1: hidden = relu(emb) @ W1 + b1  (wave covers 256 cols) ----
  for (int g = 0; g < 4; ++g) {
    const int n0 = wv * 256 + g * 64;
    v8f acc[4];
#pragma unroll
    for (int t = 0; t < 4; ++t) {
      float bias = b1[n0 + t * 16 + (lane & 15)];
#pragma unroll
      for (int e = 0; e < 8; ++e) acc[t][e] = bias;
    }
    for (int k0 = 0; k0 < 512; k0 += 32) {
      v16bf a = relu_v(load_a_bf16(emb, r0, k0, 512, lane));
#pragma unroll
      for (int t = 0; t < 4; ++t) {
        v16bf b = load_b_bf16(w1T, n0 + t * 16, k0, 512, lane);
        acc[t] = wmma_bf16(a, b, acc[t]);
      }
    }
#pragma unroll
    for (int t = 0; t < 4; ++t) {
      int col = n0 + t * 16 + (lane & 15);
#pragma unroll
      for (int e = 0; e < 8; ++e) {
        int m = e + ((lane >> 4) << 3);
        hid[m * 2048 + col] = f2bf(acc[t][e]);
      }
    }
  }
  __syncthreads();
  // ---- phase 2: out = hidden @ W2 + b2  (wave covers 64 cols, K=2048) ----
  const int n0 = wv * 64;
  v8f acc[4];
#pragma unroll
  for (int t = 0; t < 4; ++t) {
    float bias = b2[n0 + t * 16 + (lane & 15)];
#pragma unroll
    for (int e = 0; e < 8; ++e) acc[t][e] = bias;
  }
  for (int k0 = 0; k0 < 2048; k0 += 32) {
    v16bf a = load_a_bf16(hid, 0, k0, 2048, lane);   // A from LDS
#pragma unroll
    for (int t = 0; t < 4; ++t) {
      v16bf b = load_b_bf16(w2T, n0 + t * 16, k0, 2048, lane);
      acc[t] = wmma_bf16(a, b, acc[t]);
    }
  }
#pragma unroll
  for (int t = 0; t < 4; ++t) {
    int col = n0 + t * 16 + (lane & 15);
#pragma unroll
    for (int e = 0; e < 8; ++e) {
      int m = e + ((lane >> 4) << 3);
      __builtin_nontemporal_store(acc[t][e],
                                  &out_embed[(size_t)(r0 + m) * 512 + col]);
    }
  }
}

extern "C" void kernel_launch(void* const* d_in, const int* in_sizes, int n_in,
                              void* d_out, int out_size, void* d_ws, size_t ws_size,
                              hipStream_t stream) {
  const float* x      = (const float*)d_in[0];
  const float* W_enc  = (const float*)d_in[1];
  const float* b_enc  = (const float*)d_in[2];
  const float* W_ih   = (const float*)d_in[3];
  const float* W_hh   = (const float*)d_in[4];
  const float* b_lstm = (const float*)d_in[5];
  const float* W1     = (const float*)d_in[6];
  const float* b1     = (const float*)d_in[7];
  const float* W2     = (const float*)d_in[8];
  const float* b2     = (const float*)d_in[9];
  float* out = (float*)d_out;

  unsigned short* ws = (unsigned short*)d_ws;
  size_t off = 0;
  unsigned short* wencT = ws + off; off += (size_t)512 * 256;
  unsigned short* wihT  = ws + off; off += (size_t)2048 * 512;
  unsigned short* whhT  = ws + off; off += (size_t)2048 * 512;
  unsigned short* w1T   = ws + off; off += (size_t)2048 * 512;
  unsigned short* w2T   = ws + off; off += (size_t)512 * 2048;
  unsigned short* emb   = ws + off; off += (size_t)32768 * 512;
  unsigned short* zx    = ws + off; off += (size_t)32768 * 2048;
  unsigned short* h0    = ws + off; off += 32 * 512;
  unsigned short* h1    = ws + off; off += 32 * 512;
  unsigned int* cnt = (unsigned int*)(ws + off);   // off even -> 4B aligned

  // One-time weight convert+transpose to bf16 [N][K]
  k_transpose_cvt<<<(256 * 512 + 255) / 256, 256, 0, stream>>>(W_enc, wencT, 256, 512);
  k_transpose_cvt<<<(512 * 2048 + 255) / 256, 256, 0, stream>>>(W_ih, wihT, 512, 2048);
  k_transpose_cvt<<<(512 * 2048 + 255) / 256, 256, 0, stream>>>(W_hh, whhT, 512, 2048);
  k_transpose_cvt<<<(512 * 2048 + 255) / 256, 256, 0, stream>>>(W1, w1T, 512, 2048);
  k_transpose_cvt<<<(2048 * 512 + 255) / 256, 256, 0, stream>>>(W2, w2T, 2048, 512);
  k_init_state<<<64, 256, 0, stream>>>(h0, h1, cnt);

  k_encoder<<<dim3(1, 2048), 256, 0, stream>>>(x, wencT, b_enc, emb);
  k_zx<<<dim3(4, 2048), 256, 0, stream>>>(emb, wihT, b_lstm, zx);
  k_mlp<<<2048, 256, 0, stream>>>(emb, w1T, b1, w2T, b2,
                                  out + (size_t)16777216 + 32768);
  k_lstm_persistent<<<64, 32, 0, stream>>>(zx, whhT, h0, h1, out, cnt);
}
